// ENASController_26929444946567
// MI455X (gfx1250) — matrix operand order repaired and checked by reading, not compiled, not run on hardware
//
#include <hip/hip_runtime.h>
#include <hip/hip_bf16.h>
#include <cstdint>
#include <math.h>

#define B_SIZE   32768
#define H        512
#define G4       2048      // 4*H gate columns
#define L_STEPS  24
#define BR       8
#define TEMP     5.0f

#define MT       32        // rows (M) per workgroup (2 m-tiles of 16)
#define LDP      520       // LDS pitch in bf16 elems: 512 + 8 pad (16B) -> conflict-free
#define BLK      256       // 8 waves

typedef __attribute__((ext_vector_type(16))) __bf16       bf16x16;
typedef __attribute__((ext_vector_type(8)))  float        f32x8;
typedef __attribute__((ext_vector_type(4)))  unsigned int u32x4;
typedef __attribute__((ext_vector_type(8)))  int          i32x8;
typedef __attribute__((ext_vector_type(4)))  int          i32x4;

union ABFrag { bf16x16 m; u32x4 q[2]; };

__device__ __forceinline__ float sigm(float x) { return 1.0f / (1.0f + expf(-x)); }

__device__ __forceinline__ unsigned long long sm64(unsigned long long z) {
    z += 0x9e3779b97f4a7c15ull;
    z = (z ^ (z >> 30)) * 0xbf58476d1ce4e5b9ull;
    z = (z ^ (z >> 27)) * 0x94d049bb133111ebull;
    return z ^ (z >> 31);
}

// ---------------------------------------------------------------------------
// Prep: Wsum = bf16(W_ih + W_hh) (already (2048,512) N-major, K-contiguous =
// exactly the transposed B-matrix layout WMMA wants), bsum = b_ih + b_hh.
// Valid because inp == h for every scan step >= 1.
// ---------------------------------------------------------------------------
__global__ void prep_kernel(const float* __restrict__ Wih, const float* __restrict__ Whh,
                            const float* __restrict__ bih, const float* __restrict__ bhh,
                            __bf16* __restrict__ Wsum, float* __restrict__ bsum) {
    int idx = blockIdx.x * blockDim.x + threadIdx.x;
    if (idx < G4 * H) Wsum[idx] = (__bf16)(Wih[idx] + Whh[idx]);
    if (idx < G4)     bsum[idx] = bih[idx] + bhh[idx];
}

// Step 0: h==0, inp = broadcast g_emb row -> gates0 is a single 2048-vector.
__global__ void step0_gates_kernel(const float* __restrict__ Wih, const float* __restrict__ bih,
                                   const float* __restrict__ bhh, const float* __restrict__ gemb,
                                   float* __restrict__ gates0) {
    int n = blockIdx.x * blockDim.x + threadIdx.x;
    if (n >= G4) return;
    float s = bih[n] + bhh[n];
    const float* w = Wih + (size_t)n * H;
    for (int k = 0; k < H; ++k) s += w[k] * gemb[k];
    gates0[n] = s;
}

__global__ void step0_cell_kernel(const float* __restrict__ gates0,
                                  float* __restrict__ c1, __bf16* __restrict__ h1) {
    int j = blockIdx.x * blockDim.x + threadIdx.x;
    if (j >= H) return;
    float iv = gates0[j], gv = gates0[2 * H + j], ov = gates0[3 * H + j];
    float c2 = sigm(iv) * tanhf(gv);               // c0 == 0 -> forget path drops
    float h2 = sigm(ov) * tanhf(c2);
    c1[j] = c2;
    h1[j] = (__bf16)h2;
}

__global__ void bcast_kernel(const float* __restrict__ c1, const __bf16* __restrict__ h1,
                             float* __restrict__ cbuf, __bf16* __restrict__ hbuf) {
    int idx = blockIdx.x * blockDim.x + threadIdx.x;   // B*H = 16M < 2^31
    if (idx >= B_SIZE * H) return;
    int j = idx & (H - 1);
    cbuf[idx] = c1[j];
    hbuf[idx] = h1[j];
}

// ---------------------------------------------------------------------------
// Fused step kernel (l >= 1): gates = h @ Wsum.T + bsum, LSTM cell, h/c update.
// One block = 32 rows staged to LDS once via TDM (row-padded for bank spread).
// Each wave owns 4 n-blocks; per n-block it keeps 8 accumulators
// (4 gates x 2 m-tiles) so every B fragment feeds two WMMAs, and B fragments
// are double-buffered so loads for k+1 overlap the WMMAs of k.
// ---------------------------------------------------------------------------
__global__ void __launch_bounds__(BLK)
gemm_cell_kernel(__bf16* __restrict__ hbuf, float* __restrict__ cbuf,
                 const __bf16* __restrict__ Wsum, const float* __restrict__ bsum) {
    __shared__ __bf16 Ash[MT * LDP];                 // 33,280 B

    const int row0 = blockIdx.x * MT;
    const int tid  = threadIdx.x;
    const int wave = tid >> 5;
    const int lane = tid & 31;
    const int r    = lane & 15;
    const int half = lane >> 4;

#if __has_builtin(__builtin_amdgcn_tensor_load_to_lds)
    if (wave == 0) {
        // Tensor DMA descriptor: 2D tile MT x H bf16, row-major stride H,
        // LDS padding 4 DWORDs every 256 DWORDs (one 1024B row) -> pitch 1040B.
        unsigned long long ga = (unsigned long long)(uintptr_t)(hbuf + (size_t)row0 * H);
        unsigned int ldsa = (unsigned int)(unsigned long long)(uintptr_t)(void*)Ash;
        u32x4 g0;
        g0[0] = 1u;                                          // count=1, user mode
        g0[1] = ldsa;                                        // lds_addr (bytes)
        g0[2] = (unsigned int)ga;                            // global_addr[31:0]
        g0[3] = ((unsigned int)(ga >> 32) & 0x01FFFFFFu) | (2u << 30); // addr hi | type=2
        i32x8 g1;
        g1[0] = (int)((1u << 16)   |                         // data_size = 2B
                      (1u << 20)   |                         // pad_enable
                      (7u << 22)   |                         // pad_interval: 256 DWORDs
                      (3u << 25));                           // pad_amount: 4 DWORDs
        g1[1] = (int)(((unsigned)H & 0xFFFFu) << 16);        // tensor_dim0 lo16
        g1[2] = (int)((((unsigned)H >> 16) & 0xFFFFu) |
                      (((unsigned)B_SIZE & 0xFFFFu) << 16)); // dim0 hi | tensor_dim1 lo
        g1[3] = (int)((((unsigned)B_SIZE >> 16) & 0xFFFFu) |
                      ((unsigned)H << 16));                  // dim1 hi | tile_dim0=512
        g1[4] = (int)(unsigned)MT;                           // tile_dim1=32, tile_dim2=0
        g1[5] = (int)(unsigned)H;                            // tensor_dim0_stride = 512
        g1[6] = 0;
        g1[7] = 0;
        i32x4 z4 = {0, 0, 0, 0};
#if __clang_major__ >= 23
        i32x8 z8 = {0, 0, 0, 0, 0, 0, 0, 0};
        __builtin_amdgcn_tensor_load_to_lds(g0, g1, z4, z4, z8, 0);
#else
        __builtin_amdgcn_tensor_load_to_lds(g0, g1, z4, z4, 0);
#endif
        __builtin_amdgcn_s_wait_tensorcnt(0);
    }
#else
    for (int i = tid; i < MT * H; i += BLK) {
        int rr = i >> 9, kk = i & (H - 1);
        Ash[rr * LDP + kk] = hbuf[(size_t)(row0 + rr) * H + kk];
    }
#endif
    __syncthreads();

    // 32 tasks: one 16-column n-block each; both m-tiles computed per task.
    for (int task = wave; task < H / 16; task += 8) {
        const int nb = task * 16;

        f32x8 acc[8];                                 // [gate] mtile0, [4+gate] mtile1
#pragma unroll
        for (int g = 0; g < 8; ++g) acc[g] = (f32x8){};

        const __bf16* bbase[4];
#pragma unroll
        for (int g = 0; g < 4; ++g)
            bbase[g] = Wsum + (size_t)(g * H + nb + r) * H + half * 16;

        const int a0 = r * LDP;                       // m-tile 0 row in LDS
        const int a1 = (16 + r) * LDP;                // m-tile 1 row in LDS

        ABFrag Bq[2][4];                              // double-buffered B fragments
#pragma unroll
        for (int g = 0; g < 4; ++g) {                 // prologue: k = 0
            Bq[0][g].q[0] = *(const u32x4*)(bbase[g]);
            Bq[0][g].q[1] = *(const u32x4*)(bbase[g] + 8);
        }

#pragma unroll 2
        for (int kk = 0; kk < 16; ++kk) {             // 16 K-steps of 32
            const int cur = kk & 1, nxt = cur ^ 1;
            const int k0  = kk * 32;
            const int k1  = (kk + 1 < 16) ? (kk + 1) * 32 : 0;   // wrap: harmless refetch
#pragma unroll
            for (int g = 0; g < 4; ++g) {             // prefetch B(k+1) while k computes
                Bq[nxt][g].q[0] = *(const u32x4*)(bbase[g] + k1);
                Bq[nxt][g].q[1] = *(const u32x4*)(bbase[g] + k1 + 8);
            }
            ABFrag A0, A1;                            // A: K = k0+half*16 .. +15 (contiguous)
            A0.q[0] = *(const u32x4*)&Ash[a0 + k0 + half * 16];
            A0.q[1] = *(const u32x4*)&Ash[a0 + k0 + half * 16 + 8];
            A1.q[0] = *(const u32x4*)&Ash[a1 + k0 + half * 16];
            A1.q[1] = *(const u32x4*)&Ash[a1 + k0 + half * 16 + 8];
#pragma unroll
            for (int g = 0; g < 4; ++g) {
                acc[g]     = __builtin_amdgcn_wmma_f32_16x16x32_bf16(
                                 false, A0.m, false, Bq[cur][g].m, (short)0, acc[g],     false, false);
                acc[4 + g] = __builtin_amdgcn_wmma_f32_16x16x32_bf16(
                                 false, A1.m, false, Bq[cur][g].m, (short)0, acc[4 + g], false, false);
            }
        }

        // Cell: C/D layout -> lane half selects M base (0/8), VGPR i = row, r = col.
        const int j  = nb + r;
        const float bI = bsum[j], bF = bsum[H + j], bG = bsum[2 * H + j], bO = bsum[3 * H + j];
#pragma unroll
        for (int mt2 = 0; mt2 < 2; ++mt2) {
            const int mbase = row0 + mt2 * 16 + half * 8;
#pragma unroll
            for (int i = 0; i < 8; ++i) {
                const size_t off = (size_t)(mbase + i) * H + j;
                float iv = sigm(acc[mt2 * 4 + 0][i] + bI);
                float fv = sigm(acc[mt2 * 4 + 1][i] + bF);
                float gv = tanhf(acc[mt2 * 4 + 2][i] + bG);
                float ov = sigm(acc[mt2 * 4 + 3][i] + bO);
                float c2 = fv * cbuf[off] + iv * gv;
                float h2 = ov * tanhf(c2);
                cbuf[off] = c2;
                hbuf[off] = (__bf16)h2;               // safe: h tile already in LDS
            }
        }
    }
}

// ---------------------------------------------------------------------------
// Decode: logits = h @ dec_W[l].T + dec_b[l]; Gumbel-max categorical at T=5,
// untempered log_softmax for selected log-prob, tempered softmax for entropy.
// One wave per row; dec_W[l] staged in LDS (conflict-free: k = lane + 32*e).
// ---------------------------------------------------------------------------
__global__ void __launch_bounds__(BLK)
sample_kernel(const __bf16* __restrict__ hbuf, const float* __restrict__ decW,
              const float* __restrict__ decb, float* __restrict__ out, int l) {
    __shared__ float Wd[BR * H];
    __shared__ float bd[BR];
    const int tid = threadIdx.x;
    for (int i = tid; i < BR * H; i += BLK) Wd[i] = decW[(size_t)l * BR * H + i];
    if (tid < BR) bd[tid] = decb[l * BR + tid];
    __syncthreads();

    const int wave = tid >> 5, lane = tid & 31;
    const int row  = blockIdx.x * 8 + wave;

    float s[BR];
#pragma unroll
    for (int jj = 0; jj < BR; ++jj) s[jj] = 0.0f;

    const __bf16* hr = hbuf + (size_t)row * H;
    for (int k = lane; k < H; k += 32) {
        float hv = (float)hr[k];
#pragma unroll
        for (int jj = 0; jj < BR; ++jj) s[jj] += hv * Wd[jj * H + k];
    }
#pragma unroll
    for (int offm = 16; offm > 0; offm >>= 1)
#pragma unroll
        for (int jj = 0; jj < BR; ++jj) s[jj] += __shfl_xor(s[jj], offm, 32);

    if (lane == 0) {
        float logit[BR];
#pragma unroll
        for (int jj = 0; jj < BR; ++jj) logit[jj] = s[jj] + bd[jj];

        float mx = logit[0];
#pragma unroll
        for (int jj = 1; jj < BR; ++jj) mx = fmaxf(mx, logit[jj]);
        float se = 0.0f;
#pragma unroll
        for (int jj = 0; jj < BR; ++jj) se += expf(logit[jj] - mx);
        const float lse = mx + logf(se);            // untempered logsumexp

        float lt[BR];
        float mt = -1e30f;
#pragma unroll
        for (int jj = 0; jj < BR; ++jj) { lt[jj] = logit[jj] * (1.0f / TEMP); mt = fmaxf(mt, lt[jj]); }
        float st = 0.0f;
#pragma unroll
        for (int jj = 0; jj < BR; ++jj) st += expf(lt[jj] - mt);
        const float ltse = mt + logf(st);           // tempered logsumexp

        // Gumbel-max sample over tempered logits (counter-based RNG).
        unsigned long long base = ((unsigned long long)(l + 1) << 36) ^
                                  ((unsigned long long)row * 0x9e3779b97f4a7c15ull);
        int a = 0; float best = -1e30f;
#pragma unroll
        for (int jj = 0; jj < BR; ++jj) {
            unsigned long long rnd = sm64(base + (unsigned long long)jj);
            float u = ((float)(unsigned int)(rnd >> 40) + 0.5f) * (1.0f / 16777216.0f);
            float g = -logf(-logf(u));
            float v = lt[jj] + g;
            if (v > best) { best = v; a = jj; }
        }

        float ent = 0.0f;
#pragma unroll
        for (int jj = 0; jj < BR; ++jj) {
            float p = expf(lt[jj] - ltse);          // tempered prob
            ent -= p * (logit[jj] - lse);           // times untempered log-prob
        }

        const size_t BL = (size_t)B_SIZE * L_STEPS;
        out[(size_t)row * L_STEPS + l]           = (float)a;
        out[BL + (size_t)row * L_STEPS + l]      = logit[a] - lse;
        out[2 * BL + (size_t)row * L_STEPS + l]  = ent;
    }
}

// ---------------------------------------------------------------------------
extern "C" void kernel_launch(void* const* d_in, const int* in_sizes, int n_in,
                              void* d_out, int out_size, void* d_ws, size_t ws_size,
                              hipStream_t stream) {
    const float* Wih  = (const float*)d_in[1];
    const float* Whh  = (const float*)d_in[2];
    const float* bih  = (const float*)d_in[3];
    const float* bhh  = (const float*)d_in[4];
    const float* gemb = (const float*)d_in[5];
    const float* decW = (const float*)d_in[6];
    const float* decb = (const float*)d_in[7];
    float* out = (float*)d_out;

    char* ws = (char*)d_ws;
    size_t off = 0;
    auto carve = [&](size_t bytes) { char* p = ws + off; off += (bytes + 255) & ~(size_t)255; return p; };

    __bf16* hbuf   = (__bf16*)carve((size_t)B_SIZE * H * 2);   // 32 MB  bf16 hidden state
    float*  cbuf   = (float*) carve((size_t)B_SIZE * H * 4);   // 64 MB  f32 cell state
    __bf16* Wsum   = (__bf16*)carve((size_t)G4 * H * 2);       // 2 MB   fused weights
    float*  bsum   = (float*) carve((size_t)G4 * 4);
    float*  gates0 = (float*) carve((size_t)G4 * 4);
    float*  c1     = (float*) carve((size_t)H * 4);
    __bf16* h1     = (__bf16*)carve((size_t)H * 2);
    (void)ws_size; (void)in_sizes; (void)n_in; (void)out_size;

    prep_kernel<<<(G4 * H + BLK - 1) / BLK, BLK, 0, stream>>>(Wih, Whh, bih, bhh, Wsum, bsum);
    step0_gates_kernel<<<(G4 + BLK - 1) / BLK, BLK, 0, stream>>>(Wih, bih, bhh, gemb, gates0);
    step0_cell_kernel<<<(H + BLK - 1) / BLK, BLK, 0, stream>>>(gates0, c1, h1);
    bcast_kernel<<<(B_SIZE * H + BLK - 1) / BLK, BLK, 0, stream>>>(c1, h1, cbuf, hbuf);

    for (int l = 0; l < L_STEPS; ++l) {
        if (l > 0)
            gemm_cell_kernel<<<B_SIZE / MT, BLK, 0, stream>>>(hbuf, cbuf, Wsum, bsum);
        sample_kernel<<<B_SIZE / 8, BLK, 0, stream>>>(hbuf, decW, decb, out, l);
    }
}
